// CausalSelfAttention_2662879723948
// MI455X (gfx1250) — matrix operand order repaired and checked by reading, not compiled, not run on hardware
//
#include <hip/hip_runtime.h>
#include <hip/hip_bf16.h>

typedef __attribute__((ext_vector_type(16))) __bf16 v16bf;
typedef __attribute__((ext_vector_type(8)))  __bf16 v8bf;
typedef __attribute__((ext_vector_type(4)))  __bf16 v4bf;
typedef __attribute__((ext_vector_type(8)))  float  v8f;
typedef __attribute__((ext_vector_type(4)))  float  v4f;

#define B_   2
#define T_   2048
#define C_   1024
#define H_   16
#define HD_  64
#define BT_  (B_ * T_)     // 4096
#define C3_  (3 * C_)      // 3072

__device__ __forceinline__ v8f wmma_bf16(v16bf a, v16bf b, v8f c) {
    // D = A(16x32 bf16) * B(32x16 bf16) + C(16x16 f32)
    return __builtin_amdgcn_wmma_f32_16x16x32_bf16(
        /*neg_a=*/false, a, /*neg_b=*/false, b,
        /*c_mod=*/(short)0, c, /*reuse_a=*/false, /*reuse_b=*/false);
}

__device__ __forceinline__ v8f zero8() {
    v8f z;
#pragma unroll
    for (int e = 0; e < 8; ++e) z[e] = 0.0f;
    return z;
}

__device__ __forceinline__ v16bf combine16(v8bf lo, v8bf hi) {
    v16bf r;
#pragma unroll
    for (int e = 0; e < 8; ++e) { r[e] = lo[e]; r[e + 8] = hi[e]; }
    return r;
}

// ---------------------------------------------------------------- converts
__global__ void cvt_x_kernel(const float* __restrict__ x, __bf16* __restrict__ xb, int n4) {
    int i = blockIdx.x * blockDim.x + threadIdx.x;
    if (i < n4) {
        v4f f = ((const v4f*)x)[i];
        v4bf o;
#pragma unroll
        for (int e = 0; e < 4; ++e) o[e] = (__bf16)f[e];
        ((v4bf*)xb)[i] = o;
    }
}

// W [rows][cols] f32 row-major  ->  Wt [cols][rows] bf16, LDS-tiled 32x32
__global__ __launch_bounds__(256) void cvt_wt_kernel(
        const float* __restrict__ W, __bf16* __restrict__ Wt, int rows, int cols) {
    __shared__ __bf16 tile[32][33];
    const int tx = threadIdx.x & 31;
    const int ty = threadIdx.x >> 5;           // 0..7
    const int c0 = blockIdx.x * 32;
    const int r0 = blockIdx.y * 32;
#pragma unroll
    for (int yy = 0; yy < 32; yy += 8) {
        const int r = r0 + yy + ty;
        tile[yy + ty][tx] = (__bf16)W[(size_t)r * cols + (c0 + tx)];
    }
    __syncthreads();
#pragma unroll
    for (int yy = 0; yy < 32; yy += 8) {
        const int c = c0 + yy + ty;
        Wt[(size_t)c * rows + (r0 + tx)] = tile[tx][yy + ty];
    }
}

// ---------------------------------------------------------------- QKV GEMM
// 64x32 output tile per wave: 4 A-frags x 2 B-frags -> 8 WMMAs per K-chunk
// (~150 VGPRs: stays inside the direct 256-VGPR window, no MSB juggling).
__global__ __launch_bounds__(256) void qkv_gemm_kernel(
        const __bf16* __restrict__ xb, const __bf16* __restrict__ wt,
        const float* __restrict__ bias,
        __bf16* __restrict__ q, __bf16* __restrict__ k, __bf16* __restrict__ vt) {
    const int lane = threadIdx.x & 31;
    const int wave = threadIdx.x >> 5;
    const int tile = blockIdx.x * 8 + wave;
    const int ntn  = C3_ / 32;                 // 96
    const int tm   = tile / ntn;               // 0..63
    const int tn   = tile % ntn;
    const int l16  = lane & 15, hi = lane >> 4;

    const __bf16* arow[4];
    const __bf16* brow[2];
#pragma unroll
    for (int t = 0; t < 4; ++t)
        arow[t] = xb + (size_t)(tm * 64 + t * 16 + l16) * C_;
#pragma unroll
    for (int t = 0; t < 2; ++t)
        brow[t] = wt + (size_t)(tn * 32 + t * 16 + l16) * C_;

    v8f acc[4][2];
#pragma unroll
    for (int mt = 0; mt < 4; ++mt)
#pragma unroll
        for (int nt = 0; nt < 2; ++nt) acc[mt][nt] = zero8();

    for (int k0 = 0; k0 < C_; k0 += 32) {
        v16bf A[4], Bf[2];
#pragma unroll
        for (int t = 0; t < 4; ++t) {
            __builtin_prefetch(arow[t] + k0 + 256, 0, 1);
            A[t] = combine16(*(const v8bf*)(arow[t] + k0 + hi * 8),
                             *(const v8bf*)(arow[t] + k0 + 16 + hi * 8));
        }
#pragma unroll
        for (int t = 0; t < 2; ++t) {
            __builtin_prefetch(brow[t] + k0 + 256, 0, 1);
            const __bf16* bp = brow[t] + k0 + hi * 8;
            Bf[t] = combine16(*(const v8bf*)bp, *(const v8bf*)(bp + 16));
        }
#pragma unroll
        for (int mt = 0; mt < 4; ++mt)
#pragma unroll
            for (int nt = 0; nt < 2; ++nt)
                acc[mt][nt] = wmma_bf16(A[mt], Bf[nt], acc[mt][nt]);
    }

    // epilogue: 64-row tile lies inside one batch (T_ % 64 == 0) -> hoist b/tt
    const int rowbase = tm * 64;
    const int b       = rowbase / T_;
    const int ttbase  = rowbase - b * T_;
#pragma unroll
    for (int nt = 0; nt < 2; ++nt) {
        const int n  = tn * 32 + nt * 16 + l16;    // column in [0, 3C)
        const float bn = bias[n];
        const int sec = n / C_;                    // 0=q 1=k 2=v
        const int cc  = n % C_;
        const int h   = cc / HD_, d = cc % HD_;
        const size_t bh = (size_t)(b * H_ + h);
        __bf16* qk   = (sec == 0) ? q : k;
#pragma unroll
        for (int mt = 0; mt < 4; ++mt) {
#pragma unroll
            for (int i = 0; i < 8; ++i) {
                const int tt = ttbase + mt * 16 + hi * 8 + i;
                const __bf16 val = (__bf16)(acc[mt][nt][i] + bn);
                if (sec < 2) qk[(bh * T_  + tt) * HD_ + d]  = val;
                else         vt[(bh * HD_ + d)  * T_  + tt] = val;
            }
        }
    }
}

// ---------------------------------------------------------------- attention
// One wave per 32-query tile (2 M sub-tiles): K/V fragment loads are shared
// across both sub-tiles -> 16 WMMAs per 32-key tile for 8 K/V fragments.
__global__ __launch_bounds__(32) void attn_kernel(
        const __bf16* __restrict__ q, const __bf16* __restrict__ k,
        const __bf16* __restrict__ vt, __bf16* __restrict__ y) {
    __shared__ __align__(16) __bf16 plds[32 * 32];

    const int lane = threadIdx.x & 31;
    const int l16  = lane & 15, hi = lane >> 4;
    const int qt   = blockIdx.x % (T_ / 32);
    const int bh   = blockIdx.x / (T_ / 32);
    const int b    = bh / H_, h = bh % H_;
    const int q0   = qt * 32;

    const __bf16* qbase = q  + (size_t)bh * T_  * HD_;
    const __bf16* kbase = k  + (size_t)bh * T_  * HD_;
    const __bf16* vbase = vt + (size_t)bh * HD_ * T_;

    // Q fragments: 2 M sub-tiles x 2 K-chunks over HD=64
    v16bf QA[2][2];
#pragma unroll
    for (int mt = 0; mt < 2; ++mt)
#pragma unroll
        for (int c = 0; c < 2; ++c) {
            const __bf16* qp = qbase + (size_t)(q0 + mt * 16 + l16) * HD_ + c * 32 + hi * 8;
            QA[mt][c] = combine16(*(const v8bf*)qp, *(const v8bf*)(qp + 16));
        }

    float m[2][8], l[2][8];
    v8f accv[2][4];
#pragma unroll
    for (int mt = 0; mt < 2; ++mt) {
#pragma unroll
        for (int i = 0; i < 8; ++i) { m[mt][i] = -3.0e38f; l[mt][i] = 0.0f; }
#pragma unroll
        for (int j = 0; j < 4; ++j) accv[mt][j] = zero8();
    }

    const float sc = 0.125f;                   // 1/sqrt(HD)
    const int kend = q0 + 32;                  // causal: keys < kend

    for (int n0 = 0; n0 < kend; n0 += 32) {
        // ---- S = Q @ K^T : 4 16x16 tiles, K fragments shared across mt
        v8f s[2][2];
#pragma unroll
        for (int mt = 0; mt < 2; ++mt) { s[mt][0] = zero8(); s[mt][1] = zero8(); }
#pragma unroll
        for (int c = 0; c < 2; ++c) {
            const __bf16* kp0 = kbase + (size_t)(n0 + l16)      * HD_ + c * 32 + hi * 8;
            const __bf16* kp1 = kbase + (size_t)(n0 + 16 + l16) * HD_ + c * 32 + hi * 8;
            v16bf B0 = combine16(*(const v8bf*)kp0, *(const v8bf*)(kp0 + 16));
            v16bf B1 = combine16(*(const v8bf*)kp1, *(const v8bf*)(kp1 + 16));
#pragma unroll
            for (int mt = 0; mt < 2; ++mt) {
                s[mt][0] = wmma_bf16(QA[mt][c], B0, s[mt][0]);
                s[mt][1] = wmma_bf16(QA[mt][c], B1, s[mt][1]);
            }
        }

        // ---- scale + causal mask + online softmax (per M sub-tile)
#pragma unroll
        for (int mt = 0; mt < 2; ++mt) {
#pragma unroll
            for (int i = 0; i < 8; ++i) {
                const int row  = mt * 16 + hi * 8 + i;   // query = q0 + row
                const int qi   = q0 + row;
                const int key0 = n0 + l16;
                const int key1 = n0 + 16 + l16;
                float a0 = (key0 <= qi) ? s[mt][0][i] * sc : -1.0e30f;
                float a1 = (key1 <= qi) ? s[mt][1][i] * sc : -1.0e30f;
                float rowm = fmaxf(a0, a1);
#pragma unroll
                for (int msk = 1; msk < 16; msk <<= 1)   // reduce in 16-lane half
                    rowm = fmaxf(rowm, __shfl_xor(rowm, msk));
                const float mnew  = fmaxf(m[mt][i], rowm);
                const float alpha = __expf(m[mt][i] - mnew);
                const float e0 = __expf(a0 - mnew);
                const float e1 = __expf(a1 - mnew);
                float rsum = e0 + e1;
#pragma unroll
                for (int msk = 1; msk < 16; msk <<= 1)
                    rsum += __shfl_xor(rsum, msk);
                l[mt][i] = l[mt][i] * alpha + rsum;
                m[mt][i] = mnew;
#pragma unroll
                for (int j = 0; j < 4; ++j) accv[mt][j][i] *= alpha;
                plds[row * 32 + l16]      = (__bf16)e0;
                plds[row * 32 + 16 + l16] = (__bf16)e1;
            }
        }
        __syncthreads();                       // single-wave WG: cheap fence

        // ---- P: D-layout -> A-layout fragments from LDS
        v16bf PA[2];
#pragma unroll
        for (int mt = 0; mt < 2; ++mt) {
            const __bf16* pp = plds + (mt * 16 + l16) * 32 + hi * 8;
            PA[mt] = combine16(*(const v8bf*)pp, *(const v8bf*)(pp + 16));
        }
        __syncthreads();

        // ---- O += P @ V : V fragments shared across mt
#pragma unroll
        for (int j = 0; j < 4; ++j) {
            const __bf16* vp = vbase + (size_t)(j * 16 + l16) * T_ + n0 + hi * 8;
            v16bf VB = combine16(*(const v8bf*)vp, *(const v8bf*)(vp + 16));
#pragma unroll
            for (int mt = 0; mt < 2; ++mt)
                accv[mt][j] = wmma_bf16(PA[mt], VB, accv[mt][j]);
        }
    }

    // ---- epilogue: y[b, t, h*HD + d] = acc / l
#pragma unroll
    for (int mt = 0; mt < 2; ++mt) {
#pragma unroll
        for (int i = 0; i < 8; ++i) {
            const float inv = 1.0f / l[mt][i];
            const int t = q0 + mt * 16 + hi * 8 + i;
            __bf16* yp = y + ((size_t)(b * T_ + t)) * C_ + h * HD_;
#pragma unroll
            for (int j = 0; j < 4; ++j)
                yp[j * 16 + l16] = (__bf16)(accv[mt][j][i] * inv);
        }
    }
}

// ---------------------------------------------------------------- out proj
// 64x32 tile per wave. out(f32) = yb[4096x1024] @ W_proj[1024x1024] + b_proj
__global__ __launch_bounds__(256) void proj_gemm_kernel(
        const __bf16* __restrict__ yb, const __bf16* __restrict__ wpt,
        const float* __restrict__ bias, float* __restrict__ out) {
    const int lane = threadIdx.x & 31;
    const int wave = threadIdx.x >> 5;
    const int tile = blockIdx.x * 8 + wave;
    const int ntn  = C_ / 32;                  // 32
    const int tm   = tile / ntn;
    const int tn   = tile % ntn;
    const int l16  = lane & 15, hi = lane >> 4;

    const __bf16* arow[4];
    const __bf16* brow[2];
#pragma unroll
    for (int t = 0; t < 4; ++t)
        arow[t] = yb + (size_t)(tm * 64 + t * 16 + l16) * C_;
#pragma unroll
    for (int t = 0; t < 2; ++t)
        brow[t] = wpt + (size_t)(tn * 32 + t * 16 + l16) * C_;

    v8f acc[4][2];
#pragma unroll
    for (int mt = 0; mt < 4; ++mt)
#pragma unroll
        for (int nt = 0; nt < 2; ++nt) acc[mt][nt] = zero8();

    for (int k0 = 0; k0 < C_; k0 += 32) {
        v16bf A[4], Bf[2];
#pragma unroll
        for (int t = 0; t < 4; ++t) {
            __builtin_prefetch(arow[t] + k0 + 256, 0, 1);
            A[t] = combine16(*(const v8bf*)(arow[t] + k0 + hi * 8),
                             *(const v8bf*)(arow[t] + k0 + 16 + hi * 8));
        }
#pragma unroll
        for (int t = 0; t < 2; ++t) {
            __builtin_prefetch(brow[t] + k0 + 256, 0, 1);
            const __bf16* bp = brow[t] + k0 + hi * 8;
            Bf[t] = combine16(*(const v8bf*)bp, *(const v8bf*)(bp + 16));
        }
#pragma unroll
        for (int mt = 0; mt < 4; ++mt)
#pragma unroll
            for (int nt = 0; nt < 2; ++nt)
                acc[mt][nt] = wmma_bf16(A[mt], Bf[nt], acc[mt][nt]);
    }

#pragma unroll
    for (int nt = 0; nt < 2; ++nt) {
        const int n = tn * 32 + nt * 16 + l16;
        const float bn = bias[n];
#pragma unroll
        for (int mt = 0; mt < 4; ++mt) {
#pragma unroll
            for (int i = 0; i < 8; ++i) {
                const int t = tm * 64 + mt * 16 + hi * 8 + i;
                out[(size_t)t * C_ + n] = acc[mt][nt][i] + bn;
            }
        }
    }
}

// ---------------------------------------------------------------- launcher
extern "C" void kernel_launch(void* const* d_in, const int* in_sizes, int n_in,
                              void* d_out, int out_size, void* d_ws, size_t ws_size,
                              hipStream_t stream) {
    const float* x      = (const float*)d_in[0];
    const float* W_attn = (const float*)d_in[1];
    const float* b_attn = (const float*)d_in[2];
    const float* W_proj = (const float*)d_in[3];
    const float* b_proj = (const float*)d_in[4];
    float* out = (float*)d_out;

    char*  ws  = (char*)d_ws;
    size_t off = 0;
    auto take = [&](size_t bytes) -> void* {
        void* p = ws + off;
        off = (off + bytes + 255) & ~(size_t)255;
        return p;
    };
    __bf16* xb  = (__bf16*)take((size_t)BT_ * C_ * 2);            // 8 MB
    __bf16* wta = (__bf16*)take((size_t)C3_ * C_ * 2);            // 6 MB
    __bf16* wpt = (__bf16*)take((size_t)C_  * C_ * 2);            // 2 MB
    __bf16* qb  = (__bf16*)take((size_t)B_ * H_ * T_ * HD_ * 2);  // 8 MB
    __bf16* kb  = (__bf16*)take((size_t)B_ * H_ * T_ * HD_ * 2);  // 8 MB
    __bf16* vtb = (__bf16*)take((size_t)B_ * H_ * HD_ * T_ * 2);  // 8 MB
    __bf16* yb  = (__bf16*)take((size_t)BT_ * C_ * 2);            // 8 MB
    (void)ws_size; (void)in_sizes; (void)n_in; (void)out_size;

    // 1) precision conversion + LDS-tiled weight transposes
    cvt_x_kernel<<<(BT_ * C_ / 4 + 255) / 256, 256, 0, stream>>>(x, xb, BT_ * C_ / 4);
    cvt_wt_kernel<<<dim3(C3_ / 32, C_ / 32), 256, 0, stream>>>(W_attn, wta, C_, C3_);
    cvt_wt_kernel<<<dim3(C_  / 32, C_ / 32), 256, 0, stream>>>(W_proj, wpt, C_, C_);

    // 2) QKV projection: (BT/64)*(3C/32) = 6144 waves, 8 waves/block
    qkv_gemm_kernel<<<(BT_ / 64) * (C3_ / 32) / 8, 256, 0, stream>>>(
        xb, wta, b_attn, qb, kb, vtb);

    // 3) flash attention: one wave per 32-query tile
    attn_kernel<<<B_ * H_ * (T_ / 32), 32, 0, stream>>>(qb, kb, vtb, yb);

    // 4) output projection: (BT/64)*(C/32) = 2048 waves
    proj_gemm_kernel<<<(BT_ / 64) * (C_ / 32) / 8, 256, 0, stream>>>(
        yb, wpt, b_proj, out);
}
